// MultiHeadMemAttn_3350074491190
// MI455X (gfx1250) — compile-verified, matrix operand order checked by reading, not run on hardware
//
#include <hip/hip_runtime.h>
#include <hip/hip_bf16.h>

// Problem dims (fixed by the reference)
#define B_  32
#define S_  4096
#define QL  1024
#define D_  512
#define H_  8
#define HD_ 64
#define ML  64

#define LDA_PAD 16
#define LDS_LDA (D_ + LDA_PAD)   // 528 bf16 elems per staged A row

typedef __attribute__((ext_vector_type(16))) __bf16 bf16x16;
typedef __attribute__((ext_vector_type(8)))  __bf16 bf16x8;
typedef __attribute__((ext_vector_type(4)))  __bf16 bf16x4;
typedef __attribute__((ext_vector_type(8)))  float  f32x8;
typedef __attribute__((ext_vector_type(4)))  float  f32x4;
typedef __attribute__((ext_vector_type(4)))  int    i32x4;

// gfx1250 async Global->LDS copy path (ASYNCcnt), guarded so the file still
// compiles if this toolchain does not expose the builtins.
// Probe-learned signature: (int4 addrspace(1)*, int4 addrspace(3)*, Imm, Imm).
#if defined(__HIP_DEVICE_COMPILE__) &&                                         \
    __has_builtin(__builtin_amdgcn_global_load_async_to_lds_b128) &&           \
    __has_builtin(__builtin_amdgcn_s_wait_asynccnt)
#define USE_ASYNC_LDS 1
typedef __attribute__((address_space(1))) i32x4 as1_i32x4;
typedef __attribute__((address_space(3))) i32x4 as3_i32x4;
#else
#define USE_ASYNC_LDS 0
#endif

static __device__ __forceinline__ f32x8 wmma_bf16(bf16x16 a, bf16x16 b, f32x8 c) {
  // D = A(16x32 bf16) x B(32x16 bf16) + C(16x16 f32)
  return __builtin_amdgcn_wmma_f32_16x16x32_bf16(false, a, false, b, (short)0, c,
                                                 false, false);
}

// 16x32 bf16 fragment from row-major data (A, or B from a transposed source).
// CDNA5 16-bit A layout: lane<16 -> row lane,   k {0..7, 16..23}
//                        lane>=16 -> row lane-16, k {8..15, 24..31}
static __device__ __forceinline__ bf16x16 load_frag_bf16(const __bf16* __restrict__ base,
                                                         int ld, int lane) {
  const int r    = lane & 15;
  const int koff = (lane >> 4) << 3;  // 0 or 8
  const __bf16* p = base + (size_t)r * ld + koff;
  bf16x8 lo = *(const bf16x8*)p;         // k = koff .. koff+7
  bf16x8 hi = *(const bf16x8*)(p + 16);  // k = koff+16 .. koff+23
  return __builtin_shufflevector(lo, hi, 0, 1, 2, 3, 4, 5, 6, 7,
                                 8, 9, 10, 11, 12, 13, 14, 15);
}

// B-fragment from a K-major (row-major KxN) source: B[k,n] at base + k*ld + n.
static __device__ __forceinline__ bf16x16 load_frag_cm(const __bf16* __restrict__ base,
                                                       int ld, int lane) {
  const int n    = lane & 15;
  const int koff = (lane >> 4) << 3;
  const __bf16* p = base + (size_t)koff * ld + n;
  bf16x16 f;
#pragma unroll
  for (int t = 0; t < 8; ++t) f[t] = p[(size_t)t * ld];
#pragma unroll
  for (int t = 0; t < 8; ++t) f[8 + t] = p[(size_t)(16 + t) * ld];
  return f;
}

// Store a 16x16 f32 accumulator: lane holds column (lane&15),
// rows (vgpr + 8*(lane>=16)).
template <typename TOut>
static __device__ __forceinline__ void store_acc(TOut* __restrict__ base, int ld,
                                                 int lane, f32x8 acc) {
  const int col   = lane & 15;
  const int rbase = (lane >> 4) << 3;
#pragma unroll
  for (int r = 0; r < 8; ++r)
    base[(size_t)(rbase + r) * ld + col] = (TOut)acc[r];
}

// --- f32 -> bf16 stream convert (4 elems / thread) ------------------------
__global__ void cvt_bf16_kernel(const float* __restrict__ src, __bf16* __restrict__ dst) {
  size_t i = ((size_t)blockIdx.x * blockDim.x + threadIdx.x) * 4;
  f32x4 v = *(const f32x4*)(src + i);
  bf16x4 o;
  o[0] = (__bf16)v[0]; o[1] = (__bf16)v[1]; o[2] = (__bf16)v[2]; o[3] = (__bf16)v[3];
  *(bf16x4*)(dst + i) = o;
}

// --- 1) mem = mean over 64 non-overlapping rows of keyvalue, cast to bf16 --
__global__ void pool_kernel(const float* __restrict__ kv, __bf16* __restrict__ memb) {
  int idx = blockIdx.x * blockDim.x + threadIdx.x;  // [0, B_*ML*D_/4)
  int d4 = idx & (D_ / 4 - 1);
  int m  = (idx / (D_ / 4)) & (ML - 1);
  int b  = idx / (D_ / 4 * ML);
  const f32x4* p = (const f32x4*)(kv + ((size_t)b * S_ + (size_t)m * 64) * D_) + d4;
  f32x4 s = {0.f, 0.f, 0.f, 0.f};
#pragma unroll 8
  for (int t = 0; t < 64; ++t) s += p[(size_t)t * (D_ / 4)];
  bf16x4 o;
#pragma unroll
  for (int t = 0; t < 4; ++t) o[t] = (__bf16)(s[t] * (1.0f / 64.0f));
  *(bf16x4*)(memb + ((size_t)b * ML + m) * D_ + (size_t)d4 * 4) = o;
}

// --- 2) Out[MxD] = X[MxD] @ W[DxD]^T --------------------------------------
// One block = one 16-row m-tile; A slab staged once in LDS (async on gfx1250),
// 8 waves cover the 512 output columns (4 WMMA accumulators each).
template <typename TOut>
__global__ void __launch_bounds__(256)
gemm_xwT_kernel(const __bf16* __restrict__ X, const __bf16* __restrict__ Wb,
                TOut* __restrict__ Out) {
  __shared__ __bf16 As[16 * LDS_LDA];
  const int mt   = blockIdx.x;
  const int lane = threadIdx.x & 31;
  const int wv   = threadIdx.x >> 5;  // n-group 0..7 (64 cols each)
  const __bf16* Xb = X + (size_t)mt * 16 * D_;

  // Stage the 16x512 bf16 A slab (16 KB) into LDS: 1024 16-byte chunks.
#pragma unroll
  for (int c = 0; c < 4; ++c) {
    int chunk = (int)threadIdx.x + c * 256;  // 0..1023
    int r = chunk >> 6;                      // 64 chunks per row
    int k = (chunk & 63) << 3;
#if USE_ASYNC_LDS
    __builtin_amdgcn_global_load_async_to_lds_b128(
        (as1_i32x4*)(Xb + (size_t)r * D_ + k),
        (as3_i32x4*)&As[r * LDS_LDA + k], 0, 0);
#else
    *(bf16x8*)&As[r * LDS_LDA + k] = *(const bf16x8*)(Xb + (size_t)r * D_ + k);
#endif
  }
#if USE_ASYNC_LDS
  __builtin_amdgcn_s_wait_asynccnt(0);
#endif
  __syncthreads();

  const __bf16* Bbase = Wb + (size_t)wv * 64 * D_;  // rows n of W = cols of W^T
  f32x8 acc0 = {}, acc1 = {}, acc2 = {}, acc3 = {};
#pragma unroll 2
  for (int kk = 0; kk < D_; kk += 32) {
    if (kk + 32 < D_)
      __builtin_prefetch(Bbase + kk + 32, 0, 0);  // global_prefetch_b8
    bf16x16 a  = load_frag_bf16(&As[kk], LDS_LDA, lane);
    bf16x16 b0 = load_frag_bf16(Bbase + kk, D_, lane);
    bf16x16 b1 = load_frag_bf16(Bbase + (size_t)16 * D_ + kk, D_, lane);
    bf16x16 b2 = load_frag_bf16(Bbase + (size_t)32 * D_ + kk, D_, lane);
    bf16x16 b3 = load_frag_bf16(Bbase + (size_t)48 * D_ + kk, D_, lane);
    acc0 = wmma_bf16(a, b0, acc0);
    acc1 = wmma_bf16(a, b1, acc1);
    acc2 = wmma_bf16(a, b2, acc2);
    acc3 = wmma_bf16(a, b3, acc3);
  }
  TOut* obase = Out + (size_t)mt * 16 * D_ + wv * 64;
  store_acc(obase,      D_, lane, acc0);
  store_acc(obase + 16, D_, lane, acc1);
  store_acc(obase + 32, D_, lane, acc2);
  store_acc(obase + 48, D_, lane, acc3);
}

// --- 3) fused attention: one wave = one (b, head, 16-query tile) ----------
__global__ void __launch_bounds__(256)
attn_kernel(const __bf16* __restrict__ hq, const __bf16* __restrict__ hk,
            const __bf16* __restrict__ hv, __bf16* __restrict__ vec) {
  const int QT = QL / 16;  // 64 query tiles
  int wave = (int)((blockIdx.x * (unsigned)blockDim.x + threadIdx.x) >> 5);
  int lane = threadIdx.x & 31;
  int qt = wave % QT;
  int h  = (wave / QT) % H_;
  int b  = wave / (QT * H_);

  const __bf16* hqb  = hq + ((size_t)b * QL + (size_t)qt * 16) * D_ + h * HD_;
  const __bf16* hkb  = hk + (size_t)b * ML * D_ + h * HD_;
  const __bf16* hvb  = hv + (size_t)b * ML * D_ + h * HD_;
  __bf16*       vecb = vec + ((size_t)b * QL + (size_t)qt * 16) * D_ + h * HD_;

  // scores^T[j,i] = sum_d hk[j,d] * hq[i,d] ; 4 j-tiles of 16, K = HD_ = 64
  f32x8 s[4];
#pragma unroll
  for (int jt = 0; jt < 4; ++jt) {
    f32x8 acc = {};
#pragma unroll
    for (int kk = 0; kk < HD_; kk += 32) {
      bf16x16 a = load_frag_bf16(hkb + (size_t)jt * 16 * D_ + kk, D_, lane);  // A = hk
      bf16x16 q = load_frag_bf16(hqb + kk, D_, lane);  // B[k=d,n=i] = hq[i,d]
      acc = wmma_bf16(a, q, acc);
    }
    s[jt] = acc;
  }

  // softmax over j (64 keys). A lane holds 32 j's of query i = lane&15;
  // the partner lane (lane^16) holds the complementary 32 for the same i.
  const float scale = 0.125f;  // 1/sqrt(64)
  float mx = -3.0e38f;
#pragma unroll
  for (int jt = 0; jt < 4; ++jt)
#pragma unroll
    for (int r = 0; r < 8; ++r) mx = fmaxf(mx, s[jt][r]);
  mx = fmaxf(mx, __shfl_xor(mx, 16, 32));
  float sum = 0.f;
#pragma unroll
  for (int jt = 0; jt < 4; ++jt)
#pragma unroll
    for (int r = 0; r < 8; ++r) {
      float e = __expf(scale * (s[jt][r] - mx));
      s[jt][r] = e;
      sum += e;
    }
  sum += __shfl_xor(sum, 16, 32);
  const float inv = 1.0f / sum;

  // C-layout of s[] is congruent with the bf16 A-fragment layout:
  // concat(s[0],s[1]) is the A-frag of P for k=j 0..31; (s[2],s[3]) for 32..63.
  bf16x16 pa0, pa1;
#pragma unroll
  for (int t = 0; t < 8; ++t) {
    pa0[t]     = (__bf16)(s[0][t] * inv);
    pa0[8 + t] = (__bf16)(s[1][t] * inv);
    pa1[t]     = (__bf16)(s[2][t] * inv);
    pa1[8 + t] = (__bf16)(s[3][t] * inv);
  }

  // vec[i,d] = sum_j P[i,j] * hv[j,d] ; B = hv slice (K-major, ld = D_)
#pragma unroll
  for (int dt = 0; dt < 4; ++dt) {
    f32x8 acc = {};
    bf16x16 b0 = load_frag_cm(hvb + dt * 16, D_, lane);                    // j 0..31
    bf16x16 b1 = load_frag_cm(hvb + (size_t)32 * D_ + dt * 16, D_, lane);  // j 32..63
    acc = wmma_bf16(pa0, b0, acc);
    acc = wmma_bf16(pa1, b1, acc);
    store_acc(vecb + dt * 16, D_, lane, acc);
  }
}

extern "C" void kernel_launch(void* const* d_in, const int* in_sizes, int n_in,
                              void* d_out, int out_size, void* d_ws, size_t ws_size,
                              hipStream_t stream) {
  const float* query = (const float*)d_in[0];
  const float* kv    = (const float*)d_in[1];
  const float* Wq    = (const float*)d_in[2];
  const float* Wk    = (const float*)d_in[3];
  const float* Wv    = (const float*)d_in[4];
  const float* Wo    = (const float*)d_in[5];
  float* out = (float*)d_out;

  // bf16 workspace layout (~104 MB total)
  __bf16* memb = (__bf16*)d_ws;                        // [B_, ML, D_]
  __bf16* hkb  = memb + (size_t)B_ * ML * D_;          // [B_, ML, D_]
  __bf16* hvb  = hkb  + (size_t)B_ * ML * D_;          // [B_, ML, D_]
  __bf16* qb   = hvb  + (size_t)B_ * ML * D_;          // [B_, QL, D_]
  __bf16* hqb  = qb   + (size_t)B_ * QL * D_;          // [B_, QL, D_]
  __bf16* vecb = hqb  + (size_t)B_ * QL * D_;          // [B_, QL, D_]
  __bf16* wqb  = vecb + (size_t)B_ * QL * D_;          // [D_, D_] x4
  __bf16* wkb  = wqb  + (size_t)D_ * D_;
  __bf16* wvb  = wkb  + (size_t)D_ * D_;
  __bf16* wob  = wvb  + (size_t)D_ * D_;

  // 1) mean-pool keyvalue -> mem (bf16); one-time bf16 conversions
  pool_kernel<<<(B_ * ML * D_ / 4) / 256, 256, 0, stream>>>(kv, memb);
  cvt_bf16_kernel<<<((size_t)B_ * QL * D_ / 4) / 256, 256, 0, stream>>>(query, qb);
  cvt_bf16_kernel<<<(D_ * D_ / 4) / 256, 256, 0, stream>>>(Wq, wqb);
  cvt_bf16_kernel<<<(D_ * D_ / 4) / 256, 256, 0, stream>>>(Wk, wkb);
  cvt_bf16_kernel<<<(D_ * D_ / 4) / 256, 256, 0, stream>>>(Wv, wvb);
  cvt_bf16_kernel<<<(D_ * D_ / 4) / 256, 256, 0, stream>>>(Wo, wob);

  // 2) hq = query @ Wq^T   (2048 m-tiles)
  gemm_xwT_kernel<__bf16><<<B_ * QL / 16, 256, 0, stream>>>(qb, wqb, hqb);
  // 3) hk = mem @ Wk^T, hv = mem @ Wv^T   (128 m-tiles each)
  gemm_xwT_kernel<__bf16><<<B_ * ML / 16, 256, 0, stream>>>(memb, wkb, hkb);
  gemm_xwT_kernel<__bf16><<<B_ * ML / 16, 256, 0, stream>>>(memb, wvb, hvb);
  // 4) fused attention -> vec (bf16)
  {
    int waves = B_ * H_ * (QL / 16);  // 16384
    attn_kernel<<<waves / 8, 256, 0, stream>>>(hqb, hkb, hvb, vecb);
  }
  // 5) out = vec @ Wo^T  (f32 output)
  gemm_xwT_kernel<float><<<B_ * QL / 16, 256, 0, stream>>>(vecb, wob, out);
}
